// MinGRULayer_67662914781717
// MI455X (gfx1250) — matrix operand order repaired and checked by reading, not compiled, not run on hardware
//
#include <hip/hip_runtime.h>
#include <stdint.h>

// Problem constants (from reference): B=64, S=1024, I=512, H=512
#define B_  64
#define S_  1024
#define I_  512
#define H_  512
#define K_  (I_ + H_)      // 1024 (concat [x_t | h]) -- weight fragment K range
#define KC_ (K_ / 32)      // 32 k-chunks of 32 (bf16 WMMA depth); kc 0..15 = x-part, 16..31 = h-part
#define NT_ (H_ / 16)      // 32 column tiles

typedef __bf16 bf16x16 __attribute__((ext_vector_type(16)));
typedef float  f32x8   __attribute__((ext_vector_type(8)));

union FragU { uint4 q[2]; bf16x16 v; };   // 32 bytes: 16 bf16 = 8 VGPRs
union AccU  { float f[8]; f32x8  v; };    // 32 bytes: 8 f32

__device__ __forceinline__ uint16_t f2bf(float f) {
    union { float f; uint32_t u; } c; c.f = f;
    uint32_t u = c.u;
    u += 0x7fffu + ((u >> 16) & 1u);      // round-to-nearest-even
    return (uint16_t)(u >> 16);
}
__device__ __forceinline__ float bf2f(uint16_t h) {
    union { uint32_t u; float f; } c; c.u = ((uint32_t)h) << 16;
    return c.f;
}
__device__ __forceinline__ uint32_t pack2(float lo, float hi) {
    return (uint32_t)f2bf(lo) | ((uint32_t)f2bf(hi) << 16);
}

// ---------------------------------------------------------------------------
// Kernel 0: one-time weight prep. W[H][I+H] f32 -> bf16 B-fragments in the
// exact v_wmma_f32_16x16x32_bf16 register image:
//   lane l: n = l%16, half = l/16 ; uint slot v holds K = (v>=4?16:0) + half*8 + 2*(v&3), +1
// Storage: wfrag[(((gate*NT + nt)*KC + kc)*32 + lane)*8 + v]   (2 MB)
// ---------------------------------------------------------------------------
__global__ void prep_weights(const float* __restrict__ Wz, const float* __restrict__ Wh,
                             uint32_t* __restrict__ wfrag) {
    int t = blockIdx.x * blockDim.x + threadIdx.x;
    const int total = 2 * NT_ * KC_ * 32 * 8;
    if (t >= total) return;
    int v    =  t        & 7;
    int lane = (t >> 3)  & 31;
    int kc   = (t >> 8)  & 31;
    int nt   = (t >> 13) & 31;
    int gate =  t >> 18;
    int n    = nt * 16 + (lane & 15);
    int half = lane >> 4;
    int k    = kc * 32 + ((v & 4) ? 16 : 0) + half * 8 + (v & 3) * 2;
    const float* W = gate ? Wh : Wz;
    wfrag[t] = pack2(W[(size_t)n * K_ + k], W[(size_t)n * K_ + k + 1]);
}

// ---------------------------------------------------------------------------
// Kernel 1: parallel x-projection (68 GFLOP, whole device).
//   Az[r][n] = bz[n] + x[r]·Wz_x[n]   Ah[r][n] = bh[n] + x[r]·Wh_x[n]
// Packed as {bf16 Az, bf16 Ah} into the SAME uint32 slot of d_out that the
// scan will later overwrite with the f32 output -> zero extra workspace.
// Block: 32 rows of x staged in LDS (A-fragment layout); 8 waves, each wave
// one N-tile x 2 row-tiles x 2 gates (4 accumulators, K=512 = 16 WMMA chunks).
// ---------------------------------------------------------------------------
__global__ __launch_bounds__(256)
void precompute_xproj(const float* __restrict__ x,
                      const float* __restrict__ bz,
                      const float* __restrict__ bh,
                      const uint32_t* __restrict__ wfrag,
                      uint32_t* __restrict__ out32) {
    __shared__ uint32_t sA[2 * 16 * 32 * 8];   // 32 rows x 512 k, frag layout (32 KB)
    const int tid  = threadIdx.x;
    const int lane = tid & 31;
    const int wave = tid >> 5;
    const int ngrp = blockIdx.x & 3;
    const size_t r0 = (size_t)(blockIdx.x >> 2) * 32;   // 2048 row-blocks of 32
    const int nt   = ngrp * 8 + wave;

    // staging: LDS addr = tid + 256*i is exactly frag slot ((i)*32+ln)*8+v
    const int vS    = tid & 7;
    const int lnS   = (tid >> 3) & 31;
    const int mS    = lnS & 15;
    const int halfS = lnS >> 4;
    const int koff  = ((vS & 4) ? 16 : 0) + halfS * 8 + (vS & 3) * 2;

    #pragma unroll
    for (int i = 0; i < 32; ++i) {
        int kc  = i & 15;
        int mt2 = i >> 4;
        int k   = kc * 32 + koff;
        const float* xp = x + (r0 + mt2 * 16 + mS) * (size_t)I_ + k;
        sA[tid + 256 * i] = pack2(xp[0], xp[1]);
    }
    __syncthreads();

    AccU aZ0, aH0, aZ1, aH1;
    #pragma unroll
    for (int i = 0; i < 8; ++i) { aZ0.f[i]=0.f; aH0.f[i]=0.f; aZ1.f[i]=0.f; aH1.f[i]=0.f; }

    const uint4* wq = (const uint4*)wfrag;
    #pragma unroll
    for (int kc = 0; kc < 16; ++kc) {          // x-part: wfrag kc 0..15
        FragU fz, fh, a0, a1;
        size_t qz = (((size_t)0 * NT_ + nt) * KC_ + kc) * 64 + lane * 2;
        size_t qh = (((size_t)1 * NT_ + nt) * KC_ + kc) * 64 + lane * 2;
        fz.q[0] = wq[qz]; fz.q[1] = wq[qz + 1];
        fh.q[0] = wq[qh]; fh.q[1] = wq[qh + 1];
        const uint4* l0 = (const uint4*)&sA[(kc * 32 + lane) * 8];
        const uint4* l1 = (const uint4*)&sA[((16 + kc) * 32 + lane) * 8];
        a0.q[0] = l0[0]; a0.q[1] = l0[1];
        a1.q[0] = l1[0]; a1.q[1] = l1[1];
        aZ0.v = __builtin_amdgcn_wmma_f32_16x16x32_bf16(false, a0.v, false, fz.v, (short)0, aZ0.v, false, false);
        aH0.v = __builtin_amdgcn_wmma_f32_16x16x32_bf16(false, a0.v, false, fh.v, (short)0, aH0.v, false, false);
        aZ1.v = __builtin_amdgcn_wmma_f32_16x16x32_bf16(false, a1.v, false, fz.v, (short)0, aZ1.v, false, false);
        aH1.v = __builtin_amdgcn_wmma_f32_16x16x32_bf16(false, a1.v, false, fh.v, (short)0, aH1.v, false, false);
    }

    const int   nIdx  = nt * 16 + (lane & 15);
    const int   halfc = lane >> 4;
    const float bzv = bz[nIdx], bhv = bh[nIdx];
    #pragma unroll
    for (int v = 0; v < 8; ++v) {
        int m = v + 8 * halfc;
        out32[(r0 + m)      * H_ + nIdx] = pack2(aZ0.f[v] + bzv, aH0.f[v] + bhv);
        out32[(r0 + 16 + m) * H_ + nIdx] = pack2(aZ1.f[v] + bzv, aH1.f[v] + bhv);
    }
}

// ---------------------------------------------------------------------------
// Kernel 2: sequential scan. The recurrence is independent per batch row, so
// 4 workgroups (16 rows each, 1024 threads = 32 wave32 = 32 N-tiles) run with
// NO inter-WG sync: h lives in LDS in A-fragment layout; the epilogue writes
// h_new directly into the fragment image for the next step. Two s_barriers
// per step. Weights (recurrent half, 1 MB) stream from L2 every step —
// wfrag/out deliberately NOT __restrict__ so LICM cannot hoist the loads
// across the s-loop (round-1 failure mode: 256+ VGPRs + scratch spills).
// ---------------------------------------------------------------------------
__global__ __launch_bounds__(1024, 1)
void min_gru_scan(const uint32_t* wfrag,
                  uint32_t* out32,             // packed pre-acts in, f32 out
                  float* hlast) {
    __shared__ uint32_t sH[16 * 32 * 8];       // h[16][512] bf16, frag layout (16 KB)
    const int tid  = threadIdx.x;
    const int lane = tid & 31;
    const int nt   = tid >> 5;                 // wave id == N-tile
    const int b0   = blockIdx.x * 16;

    #pragma unroll
    for (int i = 0; i < 4; ++i) sH[tid + 1024 * i] = 0u;   // h0 = 0

    const int nIdx  = nt * 16 + (lane & 15);
    const int halfc = lane >> 4;

    // epilogue h-write position (k_local = nIdx): fixed per lane except m
    const int kcE = nIdx >> 5;
    const int rE  = nIdx & 31;
    const int vE  = ((rE >> 4) & 1) * 4 + ((rE & 7) >> 1);
    const int pE  = rE & 1;
    const int lnE = ((rE & 15) >> 3) * 16;     // + m
    uint16_t* sH16 = (uint16_t*)sH;

    const uint4* wq = (const uint4*)wfrag;
    const size_t zBase = ((size_t)0 * NT_ + nt) * KC_ + 16;   // recurrent half
    const size_t hBase = ((size_t)1 * NT_ + nt) * KC_ + 16;

    #pragma clang loop unroll(disable)
    for (int s = 0; s < S_; ++s) {
        asm volatile("" ::: "memory");
        __syncthreads();                       // h state visible to all waves

        AccU accZ, accH;
        #pragma unroll
        for (int i = 0; i < 8; ++i) { accZ.f[i] = 0.f; accH.f[i] = 0.f; }

        FragU aF[2], zF[2], hF[2];
        {
            const uint4* la = (const uint4*)&sH[lane * 8];
            aF[0].q[0] = la[0]; aF[0].q[1] = la[1];
            size_t qz = zBase * 64 + lane * 2;
            size_t qh = hBase * 64 + lane * 2;
            zF[0].q[0] = wq[qz]; zF[0].q[1] = wq[qz + 1];
            hF[0].q[0] = wq[qh]; hF[0].q[1] = wq[qh + 1];
        }
        #pragma unroll
        for (int kc = 0; kc < 16; ++kc) {      // K=512: h-part only
            const int cur = kc & 1, nxt = cur ^ 1;
            if (kc + 1 < 16) {                 // prefetch next chunk
                const uint4* la = (const uint4*)&sH[((kc + 1) * 32 + lane) * 8];
                aF[nxt].q[0] = la[0]; aF[nxt].q[1] = la[1];
                size_t qz = (zBase + kc + 1) * 64 + lane * 2;
                size_t qh = (hBase + kc + 1) * 64 + lane * 2;
                zF[nxt].q[0] = wq[qz]; zF[nxt].q[1] = wq[qz + 1];
                hF[nxt].q[0] = wq[qh]; hF[nxt].q[1] = wq[qh + 1];
            }
            accZ.v = __builtin_amdgcn_wmma_f32_16x16x32_bf16(
                false, aF[cur].v, false, zF[cur].v, (short)0, accZ.v, false, false);
            accH.v = __builtin_amdgcn_wmma_f32_16x16x32_bf16(
                false, aF[cur].v, false, hF[cur].v, (short)0, accH.v, false, false);
        }

        __syncthreads();                       // all sH reads done before update

        #pragma unroll
        for (int v = 0; v < 8; ++v) {
            int    m  = v + 8 * halfc;
            int    b  = b0 + m;
            size_t r  = (size_t)b * S_ + s;
            uint32_t pk = out32[r * H_ + nIdx];            // packed x pre-acts
            float az  = bf2f((uint16_t)(pk & 0xffffu));
            float ah  = bf2f((uint16_t)(pk >> 16));
            float z   = 1.0f / (1.0f + __expf(-(az + accZ.f[v])));
            float ht  = tanhf(ah + accH.f[v]);
            int   hIx = ((kcE * 32 + lnE + m) * 8 + vE) * 2 + pE;
            float hold = bf2f(sH16[hIx]);
            float hn  = (1.0f - z) * hold + z * ht;
            ((float*)out32)[r * H_ + nIdx] = hn;           // overwrite with output
            sH16[hIx] = f2bf(hn);                          // h for next step
            if (s == S_ - 1) hlast[(size_t)b * H_ + nIdx] = hn;
        }
    }
}

// ---------------------------------------------------------------------------
extern "C" void kernel_launch(void* const* d_in, const int* in_sizes, int n_in,
                              void* d_out, int out_size, void* d_ws, size_t ws_size,
                              hipStream_t stream) {
    const float* x  = (const float*)d_in[0];   // [B,S,I]
    const float* Wz = (const float*)d_in[1];   // [H,I+H]
    const float* bz = (const float*)d_in[2];   // [H]
    const float* Wh = (const float*)d_in[3];   // [H,I+H]
    const float* bh = (const float*)d_in[4];   // [H]

    uint32_t* out32 = (uint32_t*)d_out;                       // [B*S, H]
    float*    hlast = (float*)d_out + (size_t)B_ * S_ * H_;   // [B,H]

    uint32_t* wfrag = (uint32_t*)d_ws;         // 2 MB bf16 weight fragments

    prep_weights<<<(2 * NT_ * KC_ * 32 * 8) / 256, 256, 0, stream>>>(Wz, Wh, wfrag);
    precompute_xproj<<<(B_ * S_ / 32) * 4, 256, 0, stream>>>(x, bz, bh, wfrag, out32);
    min_gru_scan<<<B_ / 16, 1024, 0, stream>>>(wfrag, out32, hlast);
}
